// Seq2Seq_15238543966785
// MI455X (gfx1250) — matrix operand order repaired and checked by reading, not compiled, not run on hardware
//
#include <hip/hip_runtime.h>
#include <stdint.h>

// ---------------- problem constants ----------------
constexpr int Bn = 4096;   // batch
constexpr int Ln = 60;     // encoder length
constexpr int Tn = 60;     // decoder steps
constexpr int Hn = 256;    // hidden
constexpr int In = 6;      // input feat
constexpr int On = 6;      // output feat
constexpr int G4H = 4 * Hn;          // 1024 gate width
constexpr int KDEC = 288;            // dec_Wih K padded 262 -> 288 (multiple of 32)
constexpr int AIN = Hn + On;         // 262 attn input width

typedef __attribute__((ext_vector_type(16))) __bf16 v16bf;
typedef __attribute__((ext_vector_type(8)))  float  v8f;

// ---- async global->LDS (gfx1250 ASYNCcnt path) ----
#if defined(__has_builtin)
#  if __has_builtin(__builtin_amdgcn_global_load_async_to_lds_b128) && \
      __has_builtin(__builtin_amdgcn_s_wait_asynccnt)
#    define HAS_ASYNC_LDS 1
#  endif
#endif
#ifndef HAS_ASYNC_LDS
#  define HAS_ASYNC_LDS 0
#endif

#if HAS_ASYNC_LDS
typedef int v4i_vs __attribute__((vector_size(16)));
typedef v4i_vs __attribute__((address_space(1)))* gptr_v4;
typedef v4i_vs __attribute__((address_space(3)))* lptr_v4;
#endif

// ---------------- helpers ----------------
__device__ __forceinline__ unsigned short f32_to_bf16(float f) {
  union { float f; unsigned int u; } v; v.f = f;
  unsigned int r = v.u + 0x7FFFu + ((v.u >> 16) & 1u);
  return (unsigned short)(r >> 16);
}
__device__ __forceinline__ float bf16_to_f32(unsigned short h) {
  union { unsigned int u; float f; } v; v.u = ((unsigned int)h) << 16;
  return v.f;
}
__device__ __forceinline__ float sigf(float x) { return 1.0f / (1.0f + __expf(-x)); }

// =====================================================================
// WMMA bf16 GEMM:  C[M,N](f32)  (+)=  A[M,K](bf16,row-major) * W[N,K]^T
// Block: 256 threads = 8 waves; tile 128(M) x 128(N); K-step 32.
// Wave grid 4(m) x 2(n); each wave owns a 32x64 slab = 2x4 16x16 tiles.
// MODE 0: C = acc + bias (optional RELU).  MODE 1: C += acc.
// Requires M%128==0, N%128==0, K%32==0.
// =====================================================================
template <int MODE, int RELU>
__global__ __launch_bounds__(256) void wmma_gemm_bf16(
    const unsigned short* __restrict__ A, int lda,
    const unsigned short* __restrict__ W, int ldw,
    float* __restrict__ C, int ldc,
    int K, const float* __restrict__ bias)
{
  __shared__ __align__(16) unsigned short lA[128 * 32];
  __shared__ __align__(16) unsigned short lB[128 * 32];

  const int tid  = threadIdx.x;
  const int lane = tid & 31;
  const int wid  = tid >> 5;
  const int wm   = wid & 3;   // 0..3  -> 32-row slab
  const int wn   = wid >> 2;  // 0..1  -> 64-col slab
  const int rowBase = blockIdx.y * 128;
  const int nBase   = blockIdx.x * 128;

  v8f acc[2][4];
#pragma unroll
  for (int i = 0; i < 2; ++i)
#pragma unroll
    for (int j = 0; j < 4; ++j)
#pragma unroll
      for (int e = 0; e < 8; ++e) acc[i][j][e] = 0.0f;

  // fragment lane decode (ISA 16-bit WMMA layouts)
  const int m16 = lane & 15;
  const int kbA = (lane >> 4) << 3;   // A: 0 or 8   (k-halves {0-7,16-23}/{8-15,24-31})
  const int kbB = (lane >> 4) << 3;   // B: 0 or 8   (k-halves {0-7,16-23}/{8-15,24-31})

  // staging decode: 128 rows x 64 bytes for each of A and B
  const int rS = tid >> 1;            // 128 rows, 2 threads/row
  const int cS = (tid & 1) << 4;      // half-offset 0 or 16 (16B each)

  union Frag { v16bf v; uint4 q[2]; };

  const unsigned short* gA = A + (size_t)(rowBase + rS) * lda + cS;
  const unsigned short* gB = W + (size_t)(nBase + rS) * ldw + cS;

  for (int k0 = 0; k0 < K; k0 += 32) {
#if HAS_ASYNC_LDS
    {
      gptr_v4 ga = (gptr_v4)(gA + k0);
      gptr_v4 gb = (gptr_v4)(gB + k0);
      lptr_v4 la = (lptr_v4)&lA[rS * 32 + cS];
      lptr_v4 lb = (lptr_v4)&lB[rS * 32 + cS];
      __builtin_amdgcn_global_load_async_to_lds_b128(ga, la, 0, 0);
      __builtin_amdgcn_global_load_async_to_lds_b128(ga, la, 16, 0);
      __builtin_amdgcn_global_load_async_to_lds_b128(gb, lb, 0, 0);
      __builtin_amdgcn_global_load_async_to_lds_b128(gb, lb, 16, 0);
    }
#else
    {
      const uint4* s = reinterpret_cast<const uint4*>(gA + k0);
      uint4* d = reinterpret_cast<uint4*>(&lA[rS * 32 + cS]);
      d[0] = s[0]; d[1] = s[1];
      const uint4* s2 = reinterpret_cast<const uint4*>(gB + k0);
      uint4* d2 = reinterpret_cast<uint4*>(&lB[rS * 32 + cS]);
      d2[0] = s2[0]; d2[1] = s2[1];
    }
#endif
    if (k0 + 32 < K) {  // hint next K-slice (global_prefetch_b8)
      __builtin_prefetch(gA + k0 + 32, 0, 1);
      __builtin_prefetch(gB + k0 + 32, 0, 1);
    }
#if HAS_ASYNC_LDS
    __builtin_amdgcn_s_wait_asynccnt(0);
#endif
    __syncthreads();

    Frag a[2], b[4];
#pragma unroll
    for (int i = 0; i < 2; ++i) {
      const int r = wm * 32 + i * 16 + m16;
      a[i].q[0] = *reinterpret_cast<const uint4*>(&lA[r * 32 + kbA]);
      a[i].q[1] = *reinterpret_cast<const uint4*>(&lA[r * 32 + kbA + 16]);
    }
#pragma unroll
    for (int j = 0; j < 4; ++j) {
      const int n = wn * 64 + j * 16 + m16;
      b[j].q[0] = *reinterpret_cast<const uint4*>(&lB[n * 32 + kbB]);
      b[j].q[1] = *reinterpret_cast<const uint4*>(&lB[n * 32 + kbB + 16]);
    }
#pragma unroll
    for (int i = 0; i < 2; ++i)
#pragma unroll
      for (int j = 0; j < 4; ++j)
        acc[i][j] = __builtin_amdgcn_wmma_f32_16x16x32_bf16(
            false, a[i].v, false, b[j].v, (short)0, acc[i][j], false, false);
    __syncthreads();
  }

  // ---- epilogue (C/D layout: row = vgpr + 8*(lane>>4), col = lane&15) ----
  const int rOff = (lane >> 4) << 3;
#pragma unroll
  for (int i = 0; i < 2; ++i) {
#pragma unroll
    for (int j = 0; j < 4; ++j) {
      const int col  = nBase + wn * 64 + j * 16 + m16;
      const int row0 = rowBase + wm * 32 + i * 16 + rOff;
      float bv = (MODE == 0 && bias != nullptr) ? bias[col] : 0.0f;
#pragma unroll
      for (int v = 0; v < 8; ++v) {
        size_t idx = (size_t)(row0 + v) * ldc + col;
        float val = acc[i][j][v];
        if (MODE == 0) {
          val += bv;
          if (RELU) val = val > 0.0f ? val : 0.0f;
          C[idx] = val;
        } else {
          C[idx] += val;
        }
      }
    }
  }
}

// =====================================================================
// small utility / pointwise kernels
// =====================================================================
__global__ void k_fill_f32(float* p, float v, int n) {
  int i = blockIdx.x * blockDim.x + threadIdx.x;
  if (i < n) p[i] = v;
}
__global__ void k_fill_u16(unsigned short* p, unsigned short v, int n) {
  int i = blockIdx.x * blockDim.x + threadIdx.x;
  if (i < n) p[i] = v;
}
__global__ void k_cvt_bf16(const float* __restrict__ s, unsigned short* __restrict__ d, int n) {
  int i = blockIdx.x * blockDim.x + threadIdx.x;
  if (i < n) d[i] = f32_to_bf16(s[i]);
}
// pad/convert W[rows,scols] -> bf16 [rows,dcols] (zero pad)
__global__ void k_pad_cvt_bf16(const float* __restrict__ s, unsigned short* __restrict__ d,
                               int rows, int scols, int dcols) {
  int i = blockIdx.x * blockDim.x + threadIdx.x;
  if (i >= rows * dcols) return;
  int r = i / dcols, c = i - r * dcols;
  d[i] = (c < scols) ? f32_to_bf16(s[r * scols + c]) : (unsigned short)0;
}
__global__ void k_add2(const float* a, const float* b, float* d, int n) {
  int i = blockIdx.x * blockDim.x + threadIdx.x;
  if (i < n) d[i] = a[i] + b[i];
}
__global__ void k_go_token(float* tok) {  // [0,0,0,0,1,0]
  int i = blockIdx.x * blockDim.x + threadIdx.x;
  if (i < Bn * On) tok[i] = ((i % On) == 4) ? 1.0f : 0.0f;
}

// gates[b,n] = bias[n] + sum_i x[b,t,i] * Wih[n,i]      (K=6, grid (4, B))
__global__ __launch_bounds__(256) void k_xgate(const float* __restrict__ x,
                                               const float* __restrict__ Wih,
                                               const float* __restrict__ bias,
                                               float* __restrict__ gates, int t) {
  const int b = blockIdx.y;
  const int n = blockIdx.x * 256 + threadIdx.x;  // < 1024
  const float* xt = x + ((size_t)b * Ln + t) * In;
  float s = bias[n];
#pragma unroll
  for (int i = 0; i < In; ++i) s += xt[i] * Wih[n * In + i];
  gates[(size_t)b * G4H + n] = s;
}

// LSTM pointwise: gate order i,f,g,o
__global__ __launch_bounds__(256) void k_lstm(const float* __restrict__ gates,
                                              float* __restrict__ h, float* __restrict__ c,
                                              unsigned short* __restrict__ h_bf,
                                              unsigned short* __restrict__ enc_bf, int t) {
  int idx = blockIdx.x * blockDim.x + threadIdx.x;  // B*H
  if (idx >= Bn * Hn) return;
  int b = idx >> 8, hh = idx & 255;
  const float* g = gates + (size_t)b * G4H;
  float gi = g[hh], gf = g[Hn + hh], gg = g[2 * Hn + hh], go = g[3 * Hn + hh];
  float cn = sigf(gf) * c[idx] + sigf(gi) * tanhf(gg);
  float hn = sigf(go) * tanhf(cn);
  c[idx] = cn;
  h[idx] = hn;
  h_bf[idx] = f32_to_bf16(hn);
  if (enc_bf) enc_bf[((size_t)b * Ln + t) * Hn + hh] = f32_to_bf16(hn);
}

// logits[b,l] = attn_b[l] + h[b,:]·attn_W[l,0:256] + tok[b,:]·attn_W[l,256:262]
__global__ void k_attn(const float* __restrict__ h, const float* __restrict__ tok,
                       const float* __restrict__ aW, const float* __restrict__ ab,
                       float* __restrict__ logits) {
  int idx = blockIdx.x * blockDim.x + threadIdx.x;
  if (idx >= Bn * Ln) return;
  int b = idx / Ln, l = idx - b * Ln;
  const float* w = aW + (size_t)l * AIN;
  const float* hb = h + (size_t)b * Hn;
  float s = ab[l];
  for (int k = 0; k < Hn; ++k) s += hb[k] * w[k];
  const float* tb = tok + (size_t)b * On;
#pragma unroll
  for (int j = 0; j < On; ++j) s += tb[j] * w[Hn + j];
  logits[idx] = s;
}

// in-place softmax over L per row
__global__ void k_softmax(float* __restrict__ lg) {
  int b = blockIdx.x * blockDim.x + threadIdx.x;
  if (b >= Bn) return;
  float* p = lg + (size_t)b * Ln;
  float m = -1e30f;
  for (int l = 0; l < Ln; ++l) m = fmaxf(m, p[l]);
  float s = 0.0f;
  for (int l = 0; l < Ln; ++l) s += __expf(p[l] - m);
  float inv = 1.0f / s;
  for (int l = 0; l < Ln; ++l) p[l] = __expf(p[l] - m) * inv;
}

// build decoder input row (bf16, width 288): [ tok(6) | ctx(256) | pad ]
__global__ __launch_bounds__(288) void k_ctx_pack(const float* __restrict__ aw,
                                                  const unsigned short* __restrict__ enc_bf,
                                                  const float* __restrict__ tok,
                                                  unsigned short* __restrict__ inp_bf) {
  int b = blockIdx.x, col = threadIdx.x;
  unsigned short r = 0;
  if (col < On) {
    r = f32_to_bf16(tok[(size_t)b * On + col]);
  } else if (col < On + Hn) {
    int hh = col - On;
    const float* a = aw + (size_t)b * Ln;
    const unsigned short* e = enc_bf + (size_t)b * Ln * Hn + hh;
    float s = 0.0f;
    for (int l = 0; l < Ln; ++l) s += a[l] * bf16_to_f32(e[(size_t)l * Hn]);
    r = f32_to_bf16(s);
  }
  inp_bf[(size_t)b * KDEC + col] = r;
}

// out[b,t,o] = l2_b[o] + hid1[b,:]·l2_W[o,:]; tok = out
__global__ void k_out_head(const float* __restrict__ hid1, const float* __restrict__ W2,
                           const float* __restrict__ b2, float* __restrict__ out,
                           float* __restrict__ tok, int t) {
  int idx = blockIdx.x * blockDim.x + threadIdx.x;
  if (idx >= Bn * On) return;
  int b = idx / On, o = idx - b * On;
  const float* hb = hid1 + (size_t)b * Hn;
  const float* w = W2 + (size_t)o * Hn;
  float s = b2[o];
  for (int k = 0; k < Hn; ++k) s += hb[k] * w[k];
  out[((size_t)b * Tn + t) * On + o] = s;
  tok[idx] = s;
}

// =====================================================================
// host launcher
// =====================================================================
extern "C" void kernel_launch(void* const* d_in, const int* in_sizes, int n_in,
                              void* d_out, int out_size, void* d_ws, size_t ws_size,
                              hipStream_t stream) {
  (void)in_sizes; (void)n_in; (void)out_size; (void)ws_size;
  const float* x       = (const float*)d_in[0];
  const float* enc_Wih = (const float*)d_in[2];
  const float* enc_Whh = (const float*)d_in[3];
  const float* enc_bih = (const float*)d_in[4];
  const float* enc_bhh = (const float*)d_in[5];
  const float* attn_W  = (const float*)d_in[6];
  const float* attn_b  = (const float*)d_in[7];
  const float* dec_Wih = (const float*)d_in[8];
  const float* dec_Whh = (const float*)d_in[9];
  const float* dec_bih = (const float*)d_in[10];
  const float* dec_bhh = (const float*)d_in[11];
  const float* l1_W    = (const float*)d_in[12];
  const float* l1_b    = (const float*)d_in[13];
  const float* l2_W    = (const float*)d_in[14];
  const float* l2_b    = (const float*)d_in[15];
  float* out = (float*)d_out;

  // workspace carve-out
  char* ws = (char*)d_ws;
  size_t off = 0;
  auto carve = [&](size_t bytes) {
    size_t o = off;
    off = (off + bytes + 255) & ~(size_t)255;
    return (void*)(ws + o);
  };
  float*          h        = (float*)carve((size_t)Bn * Hn * 4);
  float*          c        = (float*)carve((size_t)Bn * Hn * 4);
  unsigned short* h_bf     = (unsigned short*)carve((size_t)Bn * Hn * 2);
  float*          gates    = (float*)carve((size_t)Bn * G4H * 4);
  float*          hid1     = (float*)carve((size_t)Bn * Hn * 4);
  float*          tok      = (float*)carve((size_t)Bn * On * 4);
  float*          aw       = (float*)carve((size_t)Bn * Ln * 4);
  unsigned short* inp_bf   = (unsigned short*)carve((size_t)Bn * KDEC * 2);
  unsigned short* enc_bf   = (unsigned short*)carve((size_t)Bn * Ln * Hn * 2);
  unsigned short* wEhh_bf  = (unsigned short*)carve((size_t)G4H * Hn * 2);
  unsigned short* wDhh_bf  = (unsigned short*)carve((size_t)G4H * Hn * 2);
  unsigned short* wDih_bf  = (unsigned short*)carve((size_t)G4H * KDEC * 2);
  unsigned short* wL1_bf   = (unsigned short*)carve((size_t)Hn * Hn * 2);
  float*          bias_enc = (float*)carve((size_t)G4H * 4);
  float*          bias_dec = (float*)carve((size_t)G4H * 4);

  auto blk = [](int n) { return (n + 255) / 256; };

  // ---- one-time setup ----
  k_cvt_bf16<<<blk(G4H * Hn), 256, 0, stream>>>(enc_Whh, wEhh_bf, G4H * Hn);
  k_cvt_bf16<<<blk(G4H * Hn), 256, 0, stream>>>(dec_Whh, wDhh_bf, G4H * Hn);
  k_cvt_bf16<<<blk(Hn * Hn), 256, 0, stream>>>(l1_W, wL1_bf, Hn * Hn);
  k_pad_cvt_bf16<<<blk(G4H * KDEC), 256, 0, stream>>>(dec_Wih, wDih_bf, G4H, In + Hn, KDEC);
  k_add2<<<blk(G4H), 256, 0, stream>>>(enc_bih, enc_bhh, bias_enc, G4H);
  k_add2<<<blk(G4H), 256, 0, stream>>>(dec_bih, dec_bhh, bias_dec, G4H);
  k_fill_f32<<<blk(Bn * Hn), 256, 0, stream>>>(h, 0.0f, Bn * Hn);
  k_fill_f32<<<blk(Bn * Hn), 256, 0, stream>>>(c, 0.0f, Bn * Hn);
  k_fill_u16<<<blk(Bn * Hn), 256, 0, stream>>>(h_bf, 0, Bn * Hn);

  dim3 gemm_gates(G4H / 128, Bn / 128);  // N=1024 -> (8,32)
  dim3 gemm_l1(Hn / 128, Bn / 128);      // N=256  -> (2,32)

  // ---- encoder ----
  for (int t = 0; t < Ln; ++t) {
    k_xgate<<<dim3(G4H / 256, Bn), 256, 0, stream>>>(x, enc_Wih, bias_enc, gates, t);
    wmma_gemm_bf16<1, 0><<<gemm_gates, 256, 0, stream>>>(h_bf, Hn, wEhh_bf, Hn, gates, G4H, Hn, nullptr);
    k_lstm<<<blk(Bn * Hn), 256, 0, stream>>>(gates, h, c, h_bf, enc_bf, t);
  }

  // ---- decoder init: c=0, tok=GO; h stays = h_enc ----
  k_fill_f32<<<blk(Bn * Hn), 256, 0, stream>>>(c, 0.0f, Bn * Hn);
  k_go_token<<<blk(Bn * On), 256, 0, stream>>>(tok);

  // ---- decoder ----
  for (int t = 0; t < Tn; ++t) {
    k_attn<<<blk(Bn * Ln), 256, 0, stream>>>(h, tok, attn_W, attn_b, aw);
    k_softmax<<<blk(Bn), 256, 0, stream>>>(aw);
    k_ctx_pack<<<Bn, 288, 0, stream>>>(aw, enc_bf, tok, inp_bf);
    wmma_gemm_bf16<0, 0><<<gemm_gates, 256, 0, stream>>>(inp_bf, KDEC, wDih_bf, KDEC, gates, G4H, KDEC, bias_dec);
    wmma_gemm_bf16<1, 0><<<gemm_gates, 256, 0, stream>>>(h_bf, Hn, wDhh_bf, Hn, gates, G4H, Hn, nullptr);
    k_lstm<<<blk(Bn * Hn), 256, 0, stream>>>(gates, h, c, h_bf, nullptr, 0);
    wmma_gemm_bf16<0, 1><<<gemm_l1, 256, 0, stream>>>(h_bf, Hn, wL1_bf, Hn, hid1, Hn, Hn, l1_b);
    k_out_head<<<blk(Bn * On), 256, 0, stream>>>(hid1, l2_W, l2_b, out, tok, t);
  }
}
